// DecoderWithLuongAttention_33045478375698
// MI455X (gfx1250) — matrix operand order repaired and checked by reading, not compiled, not run on hardware
//
#include <hip/hip_runtime.h>

#define B_  32
#define S_  64
#define T_  64
#define H_  512
#define V_  32000
#define H3_ 1536

typedef __attribute__((ext_vector_type(16))) __bf16 v16bf;
typedef __attribute__((ext_vector_type(8)))  float  v8f;

__device__ __forceinline__ unsigned short f2bf(float x) {
  unsigned int u = __float_as_uint(x);
  unsigned int r = u + 0x7FFFu + ((u >> 16) & 1u);   // round-to-nearest-even
  return (unsigned short)(r >> 16);
}

__device__ __forceinline__ float sigm(float x) { return 1.f / (1.f + __expf(-x)); }

union AB16 { v16bf v; uint4 q[2]; unsigned short u[16]; };

// ---------------------------------------------------------------------------
// Generic small GEMM: out = act( A[M,K](bf16) x W[N,K](bf16)^T + bias[N] )
// Block 256 thr = 8 waves, tile 32(M) x 64(N), 1 C-tile/wave.
// grid.x = N/64, grid.y = M/32. Used for the per-step W_cc GEMM.
// ---------------------------------------------------------------------------
__global__ void gemm_bf16(const unsigned short* __restrict__ A, int lda,
                          const unsigned short* __restrict__ W, int ldw,
                          const float* __restrict__ bias,
                          float* out_f32, long long ldc,
                          unsigned short* out_bf16, int ldcb,
                          int K, int act) {
  const int lane = threadIdx.x & 31;
  const int wave = threadIdx.x >> 5;
  const int mt = wave >> 2, nt = wave & 3;
  const int m0 = blockIdx.y * 32 + mt * 16;
  const int n0 = blockIdx.x * 64 + nt * 16;
  const int half = lane >> 4;
  const int l15 = lane & 15;

  v8f c = {};
  for (int k0 = 0; k0 < K; k0 += 32) {
    AB16 a, b;
    const unsigned short* ap = A + (size_t)(m0 + l15) * lda + k0 + half * 16;
    a.q[0] = *(const uint4*)(ap);
    a.q[1] = *(const uint4*)(ap + 8);
    const unsigned short* bp = W + (size_t)(n0 + l15) * ldw + k0 + half * 16;
    b.q[0] = *(const uint4*)(bp);
    b.q[1] = *(const uint4*)(bp + 8);
    c = __builtin_amdgcn_wmma_f32_16x16x32_bf16(false, a.v, false, b.v,
                                                (short)0, c, false, false);
  }
  const int n = n0 + l15;
  const float bval = bias[n];
#pragma unroll
  for (int i = 0; i < 8; ++i) {
    int m = m0 + i + half * 8;
    float val = c[i] + bval;
    if (act == 1) val = tanhf(val);
    if (out_f32)  out_f32[(long long)m * ldc + n] = val;
    if (out_bf16) out_bf16[(size_t)m * ldcb + n] = f2bf(val);
  }
}

// ---------------------------------------------------------------------------
// Big-tile GEMM: block 256 thr = 8 waves (4 along M x 2 along N),
// block tile 128(M) x 128(N); wave tile 32x64 = 2x4 C tiles.
// 8 WMMA per 6 fragment loads per k-step -> high compute density;
// W panel read only M/128 times total (L2-resident weights).
// permute=1: row m = t*B+b is scattered to out[b,t,:] (logits path).
// ---------------------------------------------------------------------------
__global__ void gemm_bf16_big(const unsigned short* __restrict__ A, int lda,
                              const unsigned short* __restrict__ W, int ldw,
                              const float* __restrict__ bias,
                              float* __restrict__ out, int K, int permute,
                              long long ldc) {
  const int lane = threadIdx.x & 31;
  const int wave = threadIdx.x >> 5;
  const int wm = wave & 3;
  const int wn = wave >> 2;
  const int half = lane >> 4;
  const int l15 = lane & 15;
  const int m_base = blockIdx.y * 128 + wm * 32;
  const int n_base = blockIdx.x * 128 + wn * 64;

  v8f acc[2][4] = {};
  for (int k0 = 0; k0 < K; k0 += 32) {
    AB16 a[2], b[4];
#pragma unroll
    for (int mi = 0; mi < 2; ++mi) {
      const unsigned short* ap =
          A + (size_t)(m_base + mi * 16 + l15) * lda + k0 + half * 16;
      a[mi].q[0] = *(const uint4*)(ap);
      a[mi].q[1] = *(const uint4*)(ap + 8);
    }
#pragma unroll
    for (int ni = 0; ni < 4; ++ni) {
      const unsigned short* bp =
          W + (size_t)(n_base + ni * 16 + l15) * ldw + k0 + half * 16;
      b[ni].q[0] = *(const uint4*)(bp);
      b[ni].q[1] = *(const uint4*)(bp + 8);
    }
#pragma unroll
    for (int mi = 0; mi < 2; ++mi)
#pragma unroll
      for (int ni = 0; ni < 4; ++ni)
        acc[mi][ni] = __builtin_amdgcn_wmma_f32_16x16x32_bf16(
            false, a[mi].v, false, b[ni].v, (short)0, acc[mi][ni], false, false);
  }
#pragma unroll
  for (int mi = 0; mi < 2; ++mi) {
#pragma unroll
    for (int ni = 0; ni < 4; ++ni) {
      const int n = n_base + ni * 16 + l15;
      const float bv = bias[n];
#pragma unroll
      for (int i = 0; i < 8; ++i) {
        const int m = m_base + mi * 16 + i + half * 8;
        const float val = acc[mi][ni][i] + bv;
        long long idx;
        if (permute) {                       // m = t*B + b  ->  out[b, t, n]
          const int bb = m % B_, tt = m / B_;
          idx = ((long long)bb * T_ + tt) * V_ + n;
        } else {
          idx = (long long)m * ldc + n;
        }
        out[idx] = val;
      }
    }
  }
}

// ---------------------------------------------------------------------------
// Fused GRU step: gh = h @ W_hh^T (+b_hh) with the gate nonlinearities as the
// GEMM epilogue. 8 blocks, each owns 64 j-columns. Each wave accumulates THREE
// C tiles (r/z/n slices at j, j+H, j+2H) sharing one A fragment per k-step.
// Carried-h bf16 is ping-ponged (read hbf_in, write hbf_out) to avoid the
// cross-block RAW race between A-fragment reads and epilogue writes.
// ---------------------------------------------------------------------------
__global__ void gru_step(const unsigned short* __restrict__ hbf_in,
                         const unsigned short* __restrict__ Whh,  // [3H, H]
                         const float* __restrict__ bhh,
                         const float* __restrict__ GX,            // [T*B, 3H]
                         const int* __restrict__ fra, int t,
                         float* __restrict__ h, float* __restrict__ h_new,
                         unsigned short* __restrict__ hbf_out) {
  const int lane = threadIdx.x & 31;
  const int wave = threadIdx.x >> 5;
  const int mt = wave >> 2, nt = wave & 3;
  const int m0 = mt * 16;                       // B=32 -> 2 M tiles
  const int jb = blockIdx.x * 64 + nt * 16;     // column base within H
  const int half = lane >> 4;
  const int l15 = lane & 15;

  v8f cr = {}, cz = {}, cn = {};
  for (int k0 = 0; k0 < H_; k0 += 32) {
    AB16 a, br, bz, bn;
    const unsigned short* ap = hbf_in + (size_t)(m0 + l15) * H_ + k0 + half * 16;
    a.q[0] = *(const uint4*)(ap);
    a.q[1] = *(const uint4*)(ap + 8);
    const unsigned short* bp = Whh + (size_t)(jb + l15) * H_ + k0 + half * 16;
    br.q[0] = *(const uint4*)(bp);
    br.q[1] = *(const uint4*)(bp + 8);
    bp += (size_t)H_ * H_;
    bz.q[0] = *(const uint4*)(bp);
    bz.q[1] = *(const uint4*)(bp + 8);
    bp += (size_t)H_ * H_;
    bn.q[0] = *(const uint4*)(bp);
    bn.q[1] = *(const uint4*)(bp + 8);
    cr = __builtin_amdgcn_wmma_f32_16x16x32_bf16(false, a.v, false, br.v, (short)0, cr, false, false);
    cz = __builtin_amdgcn_wmma_f32_16x16x32_bf16(false, a.v, false, bz.v, (short)0, cz, false, false);
    cn = __builtin_amdgcn_wmma_f32_16x16x32_bf16(false, a.v, false, bn.v, (short)0, cn, false, false);
  }
  const int j = jb + l15;
  const float br_ = bhh[j], bz_ = bhh[j + H_], bn_ = bhh[j + 2 * H_];
#pragma unroll
  for (int i = 0; i < 8; ++i) {
    const int b = m0 + i + half * 8;            // batch row
    const float* gx = GX + ((size_t)t * B_ + b) * H3_;
    const float r = sigm(gx[j] + cr[i] + br_);
    const float z = sigm(gx[j + H_] + cz[i] + bz_);
    const float n = tanhf(gx[j + 2 * H_] + r * (cn[i] + bn_));
    const size_t hi = (size_t)b * H_ + j;
    const float hold = h[hi];
    const float hn = (1.f - z) * n + z * hold;
    const float hc = (t < fra[b]) ? hn : hold;  // freeze finished sequences
    h_new[hi] = hn;
    h[hi] = hc;
    hbf_out[hi] = f2bf(hc);
  }
}

__global__ void f32_to_bf16_kernel(const float* __restrict__ in,
                                   unsigned short* __restrict__ out, long long n) {
  long long i = (long long)blockIdx.x * blockDim.x + threadIdx.x;
  if (i < n) out[i] = f2bf(in[i]);
}

// X[(t*B+b)*H + k] = bf16(emb[tgt[b,t], k])
__global__ void gather_embed(const float* __restrict__ emb, const int* __restrict__ tgt,
                             unsigned short* __restrict__ X) {
  long long i = (long long)blockIdx.x * blockDim.x + threadIdx.x;
  if (i >= (long long)T_ * B_ * H_) return;
  int k = (int)(i % H_);
  int b = (int)((i / H_) % B_);
  int t = (int)(i / ((long long)H_ * B_));
  int tok = tgt[b * T_ + t];
  X[i] = f2bf(emb[(size_t)tok * H_ + k]);
}

__global__ void init_h(const float* __restrict__ h0, float* __restrict__ h,
                       unsigned short* __restrict__ hbf) {
  int i = blockIdx.x * blockDim.x + threadIdx.x;
  if (i < B_ * H_) { float v = h0[i]; h[i] = v; hbf[i] = f2bf(v); }
}

// One block per batch row: scores -> masked softmax -> ctx; emits bf16 [ctx | h_new]
__global__ void attention(const float* __restrict__ enc, const unsigned char* __restrict__ mask,
                          const float* __restrict__ h_new, unsigned short* __restrict__ Acc) {
  int b = blockIdx.x;
  __shared__ float sh[H_];
  __shared__ float sc[S_];
  for (int j = threadIdx.x; j < H_; j += blockDim.x) sh[j] = h_new[b * H_ + j];
  __syncthreads();
  for (int s = threadIdx.x; s < S_; s += blockDim.x) {
    const float* e = enc + ((size_t)b * S_ + s) * H_;
    float acc = 0.f;
    for (int k = 0; k < H_; ++k) acc = fmaf(e[k], sh[k], acc);
    sc[s] = mask[b * S_ + s] ? acc : -1e30f;
  }
  __syncthreads();
  if (threadIdx.x == 0) {
    float m = -1e30f;
    for (int s = 0; s < S_; ++s) m = fmaxf(m, sc[s]);
    float sum = 0.f;
    for (int s = 0; s < S_; ++s) { float e = __expf(sc[s] - m); sc[s] = e; sum += e; }
    float inv = 1.f / sum;
    for (int s = 0; s < S_; ++s) sc[s] *= inv;
  }
  __syncthreads();
  for (int j = threadIdx.x; j < H_; j += blockDim.x) {
    float cx = 0.f;
    for (int s = 0; s < S_; ++s) cx = fmaf(sc[s], enc[((size_t)b * S_ + s) * H_ + j], cx);
    Acc[(size_t)b * (2 * H_) + j]      = f2bf(cx);
    Acc[(size_t)b * (2 * H_) + H_ + j] = f2bf(sh[j]);
  }
}

// In-place log_softmax over each row of V elements. grid = B*T rows.
__global__ void log_softmax_rows(float* __restrict__ out) {
  int row = blockIdx.x;
  float* p = out + (long long)row * V_;
  __shared__ float red[256];
  float m = -1e30f;
  for (int v = threadIdx.x; v < V_; v += 256) m = fmaxf(m, p[v]);
  red[threadIdx.x] = m; __syncthreads();
  for (int s = 128; s > 0; s >>= 1) {
    if (threadIdx.x < s) red[threadIdx.x] = fmaxf(red[threadIdx.x], red[threadIdx.x + s]);
    __syncthreads();
  }
  m = red[0]; __syncthreads();
  float sum = 0.f;
  for (int v = threadIdx.x; v < V_; v += 256) sum += __expf(p[v] - m);
  red[threadIdx.x] = sum; __syncthreads();
  for (int s = 128; s > 0; s >>= 1) {
    if (threadIdx.x < s) red[threadIdx.x] += red[threadIdx.x + s];
    __syncthreads();
  }
  float lse = m + __logf(red[0]);
  for (int v = threadIdx.x; v < V_; v += 256) p[v] -= lse;
}

extern "C" void kernel_launch(void* const* d_in, const int* in_sizes, int n_in,
                              void* d_out, int out_size, void* d_ws, size_t ws_size,
                              hipStream_t stream) {
  (void)in_sizes; (void)n_in; (void)out_size; (void)ws_size;
  const float* enc   = (const float*)d_in[0];
  const float* h0    = (const float*)d_in[1];
  const unsigned char* mask = (const unsigned char*)d_in[2];   // jax bool_: 1 byte/elem
  const int*   tgt   = (const int*)d_in[3];
  const int*   fra   = (const int*)d_in[4];
  const float* emb   = (const float*)d_in[5];
  const float* W_ih  = (const float*)d_in[6];
  const float* W_hh  = (const float*)d_in[7];
  const float* b_ih  = (const float*)d_in[8];
  const float* b_hh  = (const float*)d_in[9];
  const float* W_cc  = (const float*)d_in[10];
  const float* b_cc  = (const float*)d_in[11];
  const float* W_out = (const float*)d_in[12];
  const float* b_out = (const float*)d_in[13];
  float* out = (float*)d_out;

  char* ws = (char*)d_ws;
  size_t off = 0;
  auto alloc = [&](size_t bytes) -> void* {
    void* p = ws + off;
    off = (off + bytes + 255) & ~(size_t)255;
    return p;
  };
  unsigned short* wih_bf  = (unsigned short*)alloc((size_t)H3_ * H_ * 2);
  unsigned short* whh_bf  = (unsigned short*)alloc((size_t)H3_ * H_ * 2);
  unsigned short* wcc_bf  = (unsigned short*)alloc((size_t)H_ * 2 * H_ * 2);
  unsigned short* wout_bf = (unsigned short*)alloc((size_t)V_ * H_ * 2);
  unsigned short* xbf     = (unsigned short*)alloc((size_t)T_ * B_ * H_ * 2);
  float*          GX      = (float*)alloc((size_t)T_ * B_ * H3_ * 4);
  float*          hf      = (float*)alloc((size_t)B_ * H_ * 4);
  float*          hnew    = (float*)alloc((size_t)B_ * H_ * 4);
  unsigned short* hbf0    = (unsigned short*)alloc((size_t)B_ * H_ * 2);
  unsigned short* hbf1    = (unsigned short*)alloc((size_t)B_ * H_ * 2);
  unsigned short* accbf   = (unsigned short*)alloc((size_t)B_ * 2 * H_ * 2);
  unsigned short* ho_all  = (unsigned short*)alloc((size_t)T_ * B_ * H_ * 2);

  // --- weight conversions (recomputed every call: deterministic) ---
  {
    long long n;
    n = (long long)H3_ * H_;
    f32_to_bf16_kernel<<<(unsigned)((n + 255) / 256), 256, 0, stream>>>(W_ih, wih_bf, n);
    f32_to_bf16_kernel<<<(unsigned)((n + 255) / 256), 256, 0, stream>>>(W_hh, whh_bf, n);
    n = (long long)H_ * 2 * H_;
    f32_to_bf16_kernel<<<(unsigned)((n + 255) / 256), 256, 0, stream>>>(W_cc, wcc_bf, n);
    n = (long long)V_ * H_;
    f32_to_bf16_kernel<<<(unsigned)((n + 255) / 256), 256, 0, stream>>>(W_out, wout_bf, n);
  }
  gather_embed<<<(T_ * B_ * H_ + 255) / 256, 256, 0, stream>>>(emb, tgt, xbf);
  init_h<<<(B_ * H_ + 255) / 256, 256, 0, stream>>>(h0, hf, hbf0);

  // --- hoisted input-gate GEMM for all timesteps: GX[T*B, 3H] ---
  gemm_bf16_big<<<dim3(H3_ / 128, (T_ * B_) / 128), 256, 0, stream>>>(
      xbf, H_, wih_bf, H_, b_ih, GX, H_, 0, (long long)H3_);

  // --- sequential decode: 3 launches per step ---
  unsigned short* hb[2] = { hbf0, hbf1 };
  for (int t = 0; t < T_; ++t) {
    gru_step<<<dim3(H_ / 64, 1), 256, 0, stream>>>(
        hb[t & 1], whh_bf, b_hh, GX, fra, t, hf, hnew, hb[(t + 1) & 1]);
    attention<<<B_, 256, 0, stream>>>(enc, mask, hnew, accbf);
    gemm_bf16<<<dim3(H_ / 64, 1), 256, 0, stream>>>(
        accbf, 2 * H_, wcc_bf, 2 * H_, b_cc, nullptr, 0,
        ho_all + (size_t)t * B_ * H_, H_, 2 * H_, 1);
  }

  // --- hoisted batched logits GEMM: [T*B, H] x [V, H]^T -> out[B, T, V] ---
  gemm_bf16_big<<<dim3(V_ / 128, (T_ * B_) / 128), 256, 0, stream>>>(
      ho_all, H_, wout_bf, H_, b_out, out, H_, 1, 0);

  log_softmax_rows<<<B_ * T_, 256, 0, stream>>>(out);
}
